// unit_gcn_68350109548672
// MI455X (gfx1250) — matrix operand order repaired and checked by reading, not compiled, not run on hardware
//
#include <hip/hip_runtime.h>
#include <hip/hip_bf16.h>
#include <math.h>

typedef __attribute__((ext_vector_type(16))) _Float16 v16h;
typedef __attribute__((ext_vector_type(8)))  float    v8f;

#define N_BATCH 64
#define C_IN    64
#define C_OUT   64
#define T_LEN   256
#define V_J     17
#define K_SUB   3
#define G_GRP   8
#define TT      8                 // time steps per workgroup
#define COLS    136               // TT*V_J valid columns
#define COLS_P  144               // padded to 9 col-tiles of 16
#define ROWS    192               // 3*C_OUT
#define SM_PITCH 149              // fp32 row pitch: gcd(149%64,64)=1 -> conflict-free
#define SX_PITCH 72               // f16 col pitch (144B): conflict-free b128 frag loads

__device__ __forceinline__ bool dam_zero_row(int j) {
    return j == 3 || j == 6 || j == 7 || j == 9 || j == 10;
}

// ---------------------------------------------------------------------------
// k0: adjacency assembly (Af2[k][c][w][v]), bias pre-aggregation bagg[c][w],
//     WMMA-A-fragment-swizzled f16 weights, zero atomic accumulators.
// ---------------------------------------------------------------------------
__global__ void k0_prep(const float* __restrict__ A_base,
                        const float* __restrict__ DA,
                        const float* __restrict__ DAM,
                        const float* __restrict__ conv_w,
                        const float* __restrict__ conv_b,
                        float* __restrict__ Af2,     // [3][64][17][17] (w-major, v innermost)
                        float* __restrict__ bagg,    // [64][17]
                        _Float16* __restrict__ Wswz, // [12][2][32][16] frag order
                        float* __restrict__ sumArr,  // [64][64]
                        float* __restrict__ sumsqArr)// [64][64]
{
    const int tid = threadIdx.x;
    // A_final[k,c,v,w] = A_base[k,v,w] + DA[k,c%8,v,w] + DAM[k,c%8,v,w]*mask + 0.04
    for (int i = tid; i < K_SUB * C_OUT * V_J * V_J; i += blockDim.x) {
        int v = i % V_J, w = (i / V_J) % V_J, c = (i / (V_J * V_J)) % C_OUT, k = i / (V_J * V_J * C_OUT);
        int g = c & (G_GRP - 1);
        float msk = (dam_zero_row(v) || dam_zero_row(w) || (v >= 12 && w >= 12)) ? 0.f : 1.f;
        float val = A_base[(k * V_J + v) * V_J + w]
                  + DA[((k * G_GRP + g) * V_J + v) * V_J + w]
                  + DAM[((k * G_GRP + g) * V_J + v) * V_J + w] * msk + 0.04f;
        Af2[i] = val;  // stored as [k][c][w][v]
    }
    // bias flows linearly through the adjacency: bagg[c][w] = sum_{k,v} b[k*64+c]*Af[k,c,v,w]
    for (int i = tid; i < C_OUT * V_J; i += blockDim.x) {
        int c = i / V_J, w = i % V_J, g = c & (G_GRP - 1);
        float acc = 0.f;
        for (int k = 0; k < K_SUB; ++k) {
            float b = conv_b[k * C_OUT + c];
            for (int v = 0; v < V_J; ++v) {
                float msk = (dam_zero_row(v) || dam_zero_row(w) || (v >= 12 && w >= 12)) ? 0.f : 1.f;
                float a = A_base[(k * V_J + v) * V_J + w]
                        + DA[((k * G_GRP + g) * V_J + v) * V_J + w]
                        + DAM[((k * G_GRP + g) * V_J + v) * V_J + w] * msk + 0.04f;
                acc += b * a;
            }
        }
        bagg[i] = acc;
    }
    // Swizzle conv_w [192][64] fp32 -> f16 in a contiguous-k 16x32 A-fragment order (matches sX B loads):
    // lane L (M=L%16, group=L/16), halves hh: VGPR j=hh/2, half=hh&1:
    //   k = group*16 + hh
    for (int i = tid; i < 12 * 2 * 32 * 16; i += blockDim.x) {
        int hh = i & 15, lane = (i >> 4) & 31, kb = (i >> 9) & 1, rt = i >> 10;
        int grp = lane >> 4, j = hh >> 1, half = hh & 1;
        int kl = grp * 16 + hh;
        int M = rt * 16 + (lane & 15);
        int K = kb * 32 + kl;
        Wswz[i] = (_Float16)conv_w[M * C_IN + K];
    }
    // zero per-(n,c) accumulators (required every launch: graph replays re-atomically add)
    for (int i = tid; i < N_BATCH * C_OUT; i += blockDim.x) { sumArr[i] = 0.f; sumsqArr[i] = 0.f; }
}

// ---------------------------------------------------------------------------
// k1: fused 1x1 conv (WMMA f16->f32) + adjacency aggregation + partial sums.
// One block = one (batch n, 8-time-step tile). 256 threads = 8 waves.
// launch_bounds minWavesPerEU=4 caps VGPRs at 256/wave -> 2 blocks/WGP
// (2 x 135KB LDS of 320KB).
// ---------------------------------------------------------------------------
__global__ void __launch_bounds__(256, 4)
k1_gemm_agg(const float* __restrict__ x,
            const _Float16* __restrict__ Wswz,
            const float* __restrict__ Af2,
            const float* __restrict__ bagg,
            float* __restrict__ y,
            float* __restrict__ sumArr,
            float* __restrict__ sumsqArr)
{
    extern __shared__ char smem[];
    float*    sM = (float*)smem;                                  // [192][149] fp32
    _Float16* sX = (_Float16*)(smem + ROWS * SM_PITCH * 4);       // [144][72] f16, B-frag order

    const int tid = threadIdx.x;
    const int n   = blockIdx.x >> 5;          // /32 time-tiles
    const int t0  = (blockIdx.x & 31) * TT;   // first time step

    // ---- stage x[n, :, t0:t0+8, :] -> LDS, transposed (col-major over cin), f16 ----
    const float* xn = x + (size_t)n * C_IN * T_LEN * V_J;
    for (int i = tid; i < C_IN * COLS_P; i += 256) {
        int cin = i / COLS_P, col = i % COLS_P;
        float v = (col < COLS) ? xn[(size_t)cin * T_LEN * V_J + t0 * V_J + col] : 0.f;
        sX[col * SX_PITCH + cin] = (_Float16)v;
    }
    __syncthreads();

    // ---- GEMM: D[192][144] = W[192][64] x X[64][144], 12x9 tiles, K = 2x32 ----
    const int lane = tid & 31;
    const int wave = tid >> 5;
    const int lg   = lane >> 4;   // lane group (0/1)
    const int ll   = lane & 15;
    for (int tile = wave; tile < 12 * 9; tile += 8) {
        int rt = tile / 9, ct = tile % 9;
        v8f acc = {};
#pragma unroll
        for (int kb = 0; kb < 2; ++kb) {
            v16h a = *(const v16h*)(Wswz + (((rt * 2 + kb) * 32 + lane) << 4));
            v16h b = *(const v16h*)(sX + (ct * 16 + ll) * SX_PITCH + kb * 32 + lg * 16);
            acc = __builtin_amdgcn_wmma_f32_16x16x32_f16(false, a, false, b,
                                                         (short)0, acc, false, false);
        }
#pragma unroll
        for (int r = 0; r < 8; ++r) {   // D layout: VGPR r -> row r (lanes0-15) / r+8 (lanes16-31)
            int row = rt * 16 + lg * 8 + r;
            sM[row * SM_PITCH + ct * 16 + ll] = acc[r];
        }
    }
    __syncthreads();

    // ---- aggregation: y[c,t,w] = bagg[c,w] + sum_{k,v} m[k*64+c, t*17+v] * Af[k,c,v,w] ----
    // k-outer structure keeps peak live set ~ af[17] + o[8] (no spills, no VGPR-MSB).
    for (int pair = tid; pair < C_OUT * V_J; pair += 256) {
        int c = pair & 63;       // lane-consecutive c -> spread LDS banks
        int w = pair >> 6;
        float bg = bagg[c * V_J + w];
        float o[TT];
#pragma unroll
        for (int t = 0; t < TT; ++t) o[t] = bg;
        for (int k = 0; k < K_SUB; ++k) {
            float af[V_J];
            const float* afp = &Af2[((k * C_OUT + c) * V_J + w) * V_J];
#pragma unroll
            for (int v = 0; v < V_J; ++v) af[v] = afp[v];
            const float* mbase = &sM[(k * C_OUT + c) * SM_PITCH];
#pragma unroll
            for (int t = 0; t < TT; ++t) {
                const float* mrow = mbase + t * V_J;
                float acc = 0.f;
#pragma unroll
                for (int v = 0; v < V_J; ++v) acc += mrow[v] * af[v];
                o[t] += acc;
            }
        }
        float sy = 0.f, sy2 = 0.f;
        float* yp = &y[(((size_t)n * C_OUT + c) * T_LEN + t0) * V_J + w];
#pragma unroll
        for (int t = 0; t < TT; ++t) {
            yp[t * V_J] = o[t];
            sy += o[t]; sy2 += o[t] * o[t];
        }
        atomicAdd(&sumArr[n * C_OUT + c], sy);
        atomicAdd(&sumsqArr[n * C_OUT + c], sy2);
    }
}

// ---------------------------------------------------------------------------
// k2: SE gate + BN batch-stats -> per-(n,c) affine coefficients. One block.
//     q = s*y with s[n,c] = 1+sigmoid(g); BN stats from s*Σy, s²*Σy².
// ---------------------------------------------------------------------------
__global__ void k2_coef(const float* __restrict__ sumArr,
                        const float* __restrict__ sumsqArr,
                        const float* __restrict__ se_w,
                        const float* __restrict__ bn_g,
                        const float* __restrict__ bn_b,
                        float* __restrict__ aArr,   // [64][64]
                        float* __restrict__ bArr)   // [64]
{
    __shared__ float sP[N_BATCH * C_OUT];
    __shared__ float sS[N_BATCH * C_OUT];
    __shared__ float accT1[C_OUT];
    __shared__ float accT2[C_OUT];
    __shared__ float sGR[C_OUT];
    const int tid = threadIdx.x;
    const float invTV = 1.0f / (float)(T_LEN * V_J);
    for (int i = tid; i < N_BATCH * C_OUT; i += blockDim.x) sP[i] = sumArr[i] * invTV;
    if (tid < C_OUT) { accT1[tid] = 0.f; accT2[tid] = 0.f; }
    __syncthreads();
    float s0 = se_w[0], s1 = se_w[1], s2 = se_w[2];
    for (int i = tid; i < N_BATCH * C_OUT; i += blockDim.x) {
        int c = i & 63;
        float pm = (c > 0)  ? sP[i - 1] : 0.f;
        float pp = (c < 63) ? sP[i + 1] : 0.f;
        float g  = s0 * pm + s1 * sP[i] + s2 * pp;
        float s  = 1.f + 1.f / (1.f + expf(-g));
        sS[i] = s;
        atomicAdd(&accT1[c], s * sumArr[i]);
        atomicAdd(&accT2[c], s * s * sumsqArr[i]);
    }
    __syncthreads();
    if (tid < C_OUT) {
        const float invNTV = 1.0f / (float)(N_BATCH * T_LEN * V_J);
        float mu  = accT1[tid] * invNTV;
        float eq2 = accT2[tid] * invNTV;
        float var = eq2 - mu * mu;
        float rs  = rsqrtf(var + 1e-5f);
        sGR[tid]  = bn_g[tid] * rs;
        bArr[tid] = bn_b[tid] - bn_g[tid] * rs * mu;
    }
    __syncthreads();
    for (int i = tid; i < N_BATCH * C_OUT; i += blockDim.x)
        aArr[i] = sGR[i & 63] * sS[i];
}

// ---------------------------------------------------------------------------
// k3: out = relu(a[n,c]*y + b[c] + x). Memory-bound finish.
// ---------------------------------------------------------------------------
__global__ void k3_epilogue(const float* __restrict__ x,
                            const float* __restrict__ y,
                            const float* __restrict__ aArr,
                            const float* __restrict__ bArr,
                            float* __restrict__ out)
{
    const size_t total = (size_t)N_BATCH * C_OUT * T_LEN * V_J;
    for (size_t idx = (size_t)blockIdx.x * blockDim.x + threadIdx.x; idx < total;
         idx += (size_t)gridDim.x * blockDim.x) {
        int nc = (int)(idx / (T_LEN * V_J));
        float q = aArr[nc] * y[idx] + bArr[nc & 63] + x[idx];
        out[idx] = q > 0.f ? q : 0.f;
    }
}

// ---------------------------------------------------------------------------
extern "C" void kernel_launch(void* const* d_in, const int* in_sizes, int n_in,
                              void* d_out, int out_size, void* d_ws, size_t ws_size,
                              hipStream_t stream) {
    (void)in_sizes; (void)n_in; (void)out_size; (void)ws_size;
    const float* x       = (const float*)d_in[0];
    const float* A_base  = (const float*)d_in[1];
    const float* DA      = (const float*)d_in[2];
    const float* DAM     = (const float*)d_in[3];
    const float* conv_w  = (const float*)d_in[4];
    const float* conv_b  = (const float*)d_in[5];
    const float* se_w    = (const float*)d_in[6];
    const float* bn_g    = (const float*)d_in[7];
    const float* bn_b    = (const float*)d_in[8];
    float* out = (float*)d_out;

    // workspace layout (floats)
    float* ws = (float*)d_ws;
    float*    Af2      = ws;                         // 3*64*17*17 = 55488
    float*    bagg     = Af2 + 55488;                // 1088
    float*    sumArr   = bagg + 1088;                // 4096
    float*    sumsqArr = sumArr + 4096;              // 4096
    float*    aArr     = sumsqArr + 4096;            // 4096
    float*    bArr     = aArr + 4096;                // 64
    _Float16* Wswz     = (_Float16*)(bArr + 64);     // 12288 halves (= 6144 floats)
    float*    yBuf     = (float*)(Wswz + 12288);     // 64*64*256*17 = 17825792 (~71.3 MB)

    k0_prep<<<1, 256, 0, stream>>>(A_base, DA, DAM, conv_w, conv_b,
                                   Af2, bagg, Wswz, sumArr, sumsqArr);

    const size_t lds1 = (size_t)ROWS * SM_PITCH * 4 + (size_t)COLS_P * SX_PITCH * 2; // 135,168 B
    k1_gemm_agg<<<N_BATCH * (T_LEN / TT), 256, lds1, stream>>>(
        x, Wswz, Af2, bagg, yBuf, sumArr, sumsqArr);

    k2_coef<<<1, 256, 0, stream>>>(sumArr, sumsqArr, se_w, bn_g, bn_b, aArr, bArr);

    k3_epilogue<<<2048, 256, 0, stream>>>(x, yBuf, aArr, bArr, out);
}
